// ResidualAttentionBlock_7773890806359
// MI455X (gfx1250) — matrix operand order repaired and checked by reading, not compiled, not run on hardware
//
#include <hip/hip_runtime.h>
#include <hip/hip_bf16.h>

// ---------------------------------------------------------------------------
// Whisper-style residual attention block for MI455X (gfx1250, wave32, WMMA).
// All matmuls run through v_wmma_f32_16x16x32_bf16 (fp32 accumulate).
// LDS staging uses global_load_async_to_lds_b128 (ASYNCcnt) when available.
// ---------------------------------------------------------------------------

#define D_MODEL   1024
#define N_HEADS   16
#define HEAD_DIM  64
#define F_MLP     4096
#define BATCH     8
#define SEQ_T     448
#define SEQ_XA    1500

typedef __attribute__((ext_vector_type(16))) __bf16       v16bf;
typedef __attribute__((ext_vector_type(8)))  __bf16       v8bf;
typedef __attribute__((ext_vector_type(8)))  float        v8f;
typedef __attribute__((ext_vector_type(4)))  unsigned int u32x4;

// ---- CDNA5 async global->LDS copy (probe-guarded) ----
#if defined(__has_builtin)
#if __has_builtin(__builtin_amdgcn_global_load_async_to_lds_b128)
#define USE_ASYNC_LDS 1
#endif
#endif
#ifndef USE_ASYNC_LDS
#define USE_ASYNC_LDS 0
#endif

#if USE_ASYNC_LDS
// builtin signature (from compiler diagnostic): pointee is int __vector(4)
typedef __attribute__((__vector_size__(4 * sizeof(int)))) int async_v4i;
typedef __attribute__((address_space(1))) async_v4i* async_gptr;
typedef __attribute__((address_space(3))) async_v4i* async_lptr;

__device__ __forceinline__ void async_copy_b128(const void* g, void* l) {
  __builtin_amdgcn_global_load_async_to_lds_b128(
      (async_gptr)g, (async_lptr)l, 0, 0);
}
#define ASYNC_FENCE() asm volatile("s_wait_asynccnt 0x0" ::: "memory")
#else
#define ASYNC_FENCE()
#endif

__device__ __forceinline__ unsigned short f32_to_bf16_bits(float f) {
  unsigned int u = __builtin_bit_cast(unsigned int, f);
  u += 0x7FFFu + ((u >> 16) & 1u);          // round-to-nearest-even
  return (unsigned short)(u >> 16);
}

__device__ __forceinline__ v16bf make_frag(u32x4 lo, u32x4 hi) {
  v8bf a = __builtin_bit_cast(v8bf, lo);
  v8bf b = __builtin_bit_cast(v8bf, hi);
  return __builtin_shufflevector(a, b, 0,1,2,3,4,5,6,7,8,9,10,11,12,13,14,15);
}

__device__ __forceinline__ v8f wmma_bf(v16bf a, v16bf b, v8f c) {
  // (neg_a, A, neg_b, B, c_mod, C, reuse_a, reuse_b)
  return __builtin_amdgcn_wmma_f32_16x16x32_bf16(false, a, false, b,
                                                 (short)0, c, false, false);
}

__device__ __forceinline__ float gelu_exact(float v) {
  return 0.5f * v * (1.0f + erff(v * 0.70710678118654752f));
}

// ------------------------------- elementwise -------------------------------

__global__ void f32_to_bf16_kernel(const float* __restrict__ in,
                                   unsigned short* __restrict__ out, size_t n) {
  size_t i = (size_t)blockIdx.x * blockDim.x + threadIdx.x;
  size_t stride = (size_t)gridDim.x * blockDim.x;
  for (; i < n; i += stride) out[i] = f32_to_bf16_bits(in[i]);
}

// ------------------------------- layernorm ---------------------------------

__global__ __launch_bounds__(256)
void layernorm_kernel(const float* __restrict__ x, const float* __restrict__ g,
                      const float* __restrict__ bta,
                      unsigned short* __restrict__ out) {
  const int row = blockIdx.x;
  const float* xr = x + (size_t)row * D_MODEL;
  const int tid = threadIdx.x;
  float vals[4], s = 0.f, s2 = 0.f;
#pragma unroll
  for (int i = 0; i < 4; ++i) {
    float v = xr[tid + i * 256];
    vals[i] = v; s += v; s2 += v * v;
  }
#pragma unroll
  for (int off = 16; off; off >>= 1) {
    s  += __shfl_xor(s,  off, 32);
    s2 += __shfl_xor(s2, off, 32);
  }
  __shared__ float ps[8], ps2[8];
  const int lane = tid & 31, wid = tid >> 5;
  if (lane == 0) { ps[wid] = s; ps2[wid] = s2; }
  __syncthreads();
  if (wid == 0) {
    s  = (lane < 8) ? ps[lane]  : 0.f;
    s2 = (lane < 8) ? ps2[lane] : 0.f;
#pragma unroll
    for (int off = 4; off; off >>= 1) {
      s  += __shfl_xor(s,  off, 32);
      s2 += __shfl_xor(s2, off, 32);
    }
    if (lane == 0) { ps[0] = s; ps2[0] = s2; }
  }
  __syncthreads();
  const float mean = ps[0] * (1.0f / D_MODEL);
  const float var  = ps2[0] * (1.0f / D_MODEL) - mean * mean;
  const float rstd = rsqrtf(var + 1e-5f);
#pragma unroll
  for (int i = 0; i < 4; ++i) {
    int c = tid + i * 256;
    float v = (vals[i] - mean) * rstd * g[c] + bta[c];
    out[(size_t)row * D_MODEL + c] = f32_to_bf16_bits(v);
  }
}

// --------------------------------- GEMM ------------------------------------
// C[M,N] = A[M,K](bf16) * B[K,N](bf16) (+bias), compile-time epilogue.
// 256 thr / 8 waves; macro tile 64x128, BK=32; wave = 32x32 (4 wmma / K-step).

enum { EPI_F32 = 1, EPI_B16 = 2, EPI_RES = 4, EPI_GELU = 8 };

#define GBM 64
#define GBN 128
#define GBK 32
#define GLDA 40   // padded LDS strides (elements)
#define GLDB 40

template <int EPI>
__global__ __launch_bounds__(256)
void gemm_bf16_kernel(const unsigned short* __restrict__ A,
                      const unsigned short* __restrict__ Bw,
                      const float* __restrict__ bias,
                      const float* __restrict__ res,
                      float* __restrict__ outF,
                      unsigned short* __restrict__ outB,
                      int Mrows, int N, int K) {
  __shared__ __align__(16) unsigned short As[GBM * GLDA];
  __shared__ __align__(16) unsigned short Bs[GBN * GLDB];

  const int tid = threadIdx.x;
  const int lane = tid & 31, wid = tid >> 5;
  const int hf = lane >> 4, ln = lane & 15;
  const int wr = wid & 1, wc = wid >> 1;        // waves: 2 (M) x 4 (N)
  const int m0 = blockIdx.x * GBM;
  const int n0 = blockIdx.y * GBN;

  v8f acc[2][2] = {};

  for (int k0 = 0; k0 < K; k0 += GBK) {
    // warm L2 for the next B tile while we stage this one
    if (k0 + GBK < K)
      __builtin_prefetch(Bw + (size_t)(k0 + GBK + (tid >> 4)) * N + n0 + (tid & 15) * 8, 0, 0);

    // --- stage A tile (row major, 16B per lane) ---
    {
      int row = tid >> 2;             // 0..63
      int col = (tid & 3) * 8;        // 0,8,16,24
      int gm = m0 + row;
#if USE_ASYNC_LDS
      if (gm < Mrows)                 // OOB rows feed only discarded outputs
        async_copy_b128(A + (size_t)gm * K + k0 + col, &As[row * GLDA + col]);
#else
      u32x4 d = (u32x4)0u;
      if (gm < Mrows)
        d = *reinterpret_cast<const u32x4*>(A + (size_t)gm * K + k0 + col);
      *reinterpret_cast<u32x4*>(&As[row * GLDA + col]) = d;
#endif
    }
    // --- stage B tile transposed: Bs[n][k] ---
#pragma unroll
    for (int it = 0; it < 2; ++it) {
      int kk = (tid >> 4) + it * 16;  // 0..31
      int nn = (tid & 15) * 8;        // 0..120
      u32x4 d = *reinterpret_cast<const u32x4*>(Bw + (size_t)(k0 + kk) * N + n0 + nn);
      const unsigned short* s = reinterpret_cast<const unsigned short*>(&d);
#pragma unroll
      for (int j = 0; j < 8; ++j) Bs[(nn + j) * GLDB + kk] = s[j];
    }
    ASYNC_FENCE();
    __syncthreads();

    v16bf afr[2], bfr[2];
#pragma unroll
    for (int i = 0; i < 2; ++i) {   // A-layout: j<8 -> K=8*hf+j ; j>=8 -> K=16+8*hf+(j-8)
      const unsigned short* p = &As[(wr * 32 + i * 16 + ln) * GLDA];
      u32x4 lo = *reinterpret_cast<const u32x4*>(p + 8 * hf);
      u32x4 hi = *reinterpret_cast<const u32x4*>(p + 16 + 8 * hf);
      afr[i] = make_frag(lo, hi);
    }
#pragma unroll
    for (int j = 0; j < 2; ++j) {   // B-layout: e<8 -> K=8*hf+e ; e>=8 -> K=16+8*hf+(e-8), N = ln
      const unsigned short* p = &Bs[(wc * 32 + j * 16 + ln) * GLDB];
      u32x4 lo = *reinterpret_cast<const u32x4*>(p + 8 * hf);
      u32x4 hi = *reinterpret_cast<const u32x4*>(p + 16 + 8 * hf);
      bfr[j] = make_frag(lo, hi);
    }
#pragma unroll
    for (int i = 0; i < 2; ++i)
#pragma unroll
      for (int j = 0; j < 2; ++j)
        acc[i][j] = wmma_bf(afr[i], bfr[j], acc[i][j]);
    __syncthreads();
  }

  // --- epilogue (D layout: M = r + 8*hf, N = ln); straight-line per EPI ---
#pragma unroll
  for (int i = 0; i < 2; ++i) {
    int gmb = m0 + wr * 32 + i * 16 + 8 * hf;
#pragma unroll
    for (int j = 0; j < 2; ++j) {
      int gn = n0 + wc * 32 + j * 16 + ln;
      float bb = bias ? bias[gn] : 0.f;
#pragma unroll
      for (int r = 0; r < 8; ++r) {
        int gm = gmb + r;
        if (gm >= Mrows) continue;
        size_t idx = (size_t)gm * N + gn;
        float v = acc[i][j][r] + bb;
        if constexpr (EPI & EPI_GELU) v = gelu_exact(v);
        if constexpr (EPI & EPI_RES)  v += res[idx];
        if constexpr (EPI & EPI_F32)  outF[idx] = v;
        if constexpr (EPI & EPI_B16)  outB[idx] = f32_to_bf16_bits(v);
      }
    }
  }
}

// ----------------------------- flash attention -----------------------------
// grid: (ceil(T/128), H, B); 8 waves * 16 q-rows = 128 rows per block.
// KV tiles of 32. S = Q*K^T : 4 wmma/wave/iter ; O += P*V : 4 wmma/wave/iter.

__global__ __launch_bounds__(256)
void attn_kernel(const unsigned short* __restrict__ Q,
                 const unsigned short* __restrict__ Kb,
                 const unsigned short* __restrict__ Vb,
                 unsigned short* __restrict__ O,
                 int T, int Lk, int causal, float scale) {
  const int h = blockIdx.y, b = blockIdx.z;
  const int qbase = blockIdx.x * 128;
  const int tid = threadIdx.x;
  const int lane = tid & 31, wid = tid >> 5;
  const int hf = lane >> 4, ln = lane & 15;
  const int H64 = h * HEAD_DIM;

  __shared__ __align__(16) unsigned short Ks[32 * 72];      // [l][d]
  __shared__ __align__(16) unsigned short Vt[64 * 40];      // [d][l] (transposed)
  __shared__ __align__(16) unsigned short Ps[8][16 * 40];   // per-wave P staging

  // --- Q fragments (A layout), 16 rows per wave, hd=64 -> 2 K-chunks ---
  const int tq = qbase + wid * 16 + ln;
  v16bf qf[2];
#pragma unroll
  for (int c = 0; c < 2; ++c) {
    u32x4 lo = (u32x4)0u, hi = (u32x4)0u;
    if (tq < T) {
      const unsigned short* qp =
          Q + ((size_t)b * T + tq) * D_MODEL + H64 + c * 32;
      lo = *reinterpret_cast<const u32x4*>(qp + 8 * hf);
      hi = *reinterpret_cast<const u32x4*>(qp + 16 + 8 * hf);
    }
    qf[c] = make_frag(lo, hi);
  }

  v8f o_acc[4] = {};
  float m_r[8], l_r[8];
#pragma unroll
  for (int r = 0; r < 8; ++r) { m_r[r] = -1e30f; l_r[r] = 0.f; }

  int lend = Lk;
  if (causal) {
    int tmax = (T - 1 < qbase + 127) ? (T - 1) : (qbase + 127);
    lend = (Lk < tmax + 1) ? Lk : (tmax + 1);
  }
  const int trow_base = qbase + wid * 16 + 8 * hf;

  for (int l0 = 0; l0 < lend; l0 += 32) {
    // --- stage K (async, row major) and V (transposed, manual) tiles ---
    {
      int row = tid >> 3;            // 0..31  (kv row in tile)
      int col = (tid & 7) * 8;       // 0..56  (d)
      int gl = l0 + row;
      u32x4 dv = (u32x4)0u;
      if (gl < Lk) {
        size_t base = ((size_t)b * Lk + gl) * D_MODEL + H64 + col;
#if USE_ASYNC_LDS
        async_copy_b128(Kb + base, &Ks[row * 72 + col]);
#else
        *reinterpret_cast<u32x4*>(&Ks[row * 72 + col]) =
            *reinterpret_cast<const u32x4*>(Kb + base);
#endif
        dv = *reinterpret_cast<const u32x4*>(Vb + base);
      } else {
        *reinterpret_cast<u32x4*>(&Ks[row * 72 + col]) = (u32x4)0u;
      }
      const unsigned short* s = reinterpret_cast<const unsigned short*>(&dv);
#pragma unroll
      for (int j = 0; j < 8; ++j) Vt[(col + j) * 40 + row] = s[j];
    }
    ASYNC_FENCE();
    __syncthreads();

    // --- S = Q * K^T  (two 16-col tiles) ---
    v8f s_acc[2] = {};
#pragma unroll
    for (int jt = 0; jt < 2; ++jt) {
      const unsigned short* kp = &Ks[(jt * 16 + ln) * 72];
#pragma unroll
      for (int c = 0; c < 2; ++c) {
        u32x4 lo = *reinterpret_cast<const u32x4*>(kp + c * 32 + 8 * hf);
        u32x4 hi = *reinterpret_cast<const u32x4*>(kp + c * 32 + 16 + 8 * hf);
        s_acc[jt] = wmma_bf(qf[c], make_frag(lo, hi), s_acc[jt]);
      }
    }

    // --- online softmax over this lane's 8 rows (rows r+8*hf) ---
    float sv[2][8], rowmax[8];
    const int lc0 = l0 + ln, lc1 = l0 + 16 + ln;
#pragma unroll
    for (int r = 0; r < 8; ++r) rowmax[r] = -1e30f;
#pragma unroll
    for (int jt = 0; jt < 2; ++jt) {
      int lc = jt ? lc1 : lc0;
#pragma unroll
      for (int r = 0; r < 8; ++r) {
        float s = s_acc[jt][r] * scale;
        int tr = trow_base + r;
        bool masked = (lc >= Lk) || (causal && lc > tr);
        s = masked ? -1e30f : s;
        sv[jt][r] = s;
        rowmax[r] = fmaxf(rowmax[r], s);
      }
    }
#pragma unroll
    for (int off = 8; off; off >>= 1)
#pragma unroll
      for (int r = 0; r < 8; ++r)
        rowmax[r] = fmaxf(rowmax[r], __shfl_xor(rowmax[r], off, 16));

    float alpha[8], rowsum[8];
#pragma unroll
    for (int r = 0; r < 8; ++r) {
      float mnew = fmaxf(m_r[r], rowmax[r]);
      alpha[r] = __expf(m_r[r] - mnew);
      m_r[r] = mnew;
      rowsum[r] = 0.f;
    }
#pragma unroll
    for (int jt = 0; jt < 2; ++jt)
#pragma unroll
      for (int r = 0; r < 8; ++r) {
        float p = (sv[jt][r] <= -1e29f) ? 0.f : __expf(sv[jt][r] - m_r[r]);
        rowsum[r] += p;
        Ps[wid][(r + 8 * hf) * 40 + jt * 16 + ln] = f32_to_bf16_bits(p);
      }
#pragma unroll
    for (int off = 8; off; off >>= 1)
#pragma unroll
      for (int r = 0; r < 8; ++r) rowsum[r] += __shfl_xor(rowsum[r], off, 16);
#pragma unroll
    for (int r = 0; r < 8; ++r) l_r[r] = l_r[r] * alpha[r] + rowsum[r];
#pragma unroll
    for (int dt = 0; dt < 4; ++dt)
#pragma unroll
      for (int r = 0; r < 8; ++r) o_acc[dt][r] *= alpha[r];

    // per-wave LDS fence: D-layout P stores -> A-layout P loads (in-order DS)
    asm volatile("s_wait_dscnt 0x0" ::: "memory");

    const unsigned short* pp = &Ps[wid][ln * 40];
    u32x4 plo = *reinterpret_cast<const u32x4*>(pp + 8 * hf);
    u32x4 phi = *reinterpret_cast<const u32x4*>(pp + 16 + 8 * hf);
    v16bf pf = make_frag(plo, phi);

    // --- O += P * V  (4 d-tiles of 16) ---
#pragma unroll
    for (int dt = 0; dt < 4; ++dt) {
      const unsigned short* vp = &Vt[(dt * 16 + ln) * 40];
      u32x4 lo = *reinterpret_cast<const u32x4*>(vp + 8 * hf);
      u32x4 hi = *reinterpret_cast<const u32x4*>(vp + 16 + 8 * hf);
      o_acc[dt] = wmma_bf(pf, make_frag(lo, hi), o_acc[dt]);
    }
    __syncthreads();
  }

  float rden[8];
#pragma unroll
  for (int r = 0; r < 8; ++r) rden[r] = (l_r[r] > 0.f) ? 1.f / l_r[r] : 0.f;
#pragma unroll
  for (int dt = 0; dt < 4; ++dt)
#pragma unroll
    for (int r = 0; r < 8; ++r) {
      int t = trow_base + r;
      if (t < T) {
        float v = o_acc[dt][r] * rden[r];
        O[((size_t)b * T + t) * D_MODEL + H64 + dt * 16 + ln] =
            f32_to_bf16_bits(v);
      }
    }
}

// ------------------------------- host side ---------------------------------

static inline size_t alignup(size_t x) { return (x + 255) & ~(size_t)255; }

extern "C" void kernel_launch(void* const* d_in, const int* in_sizes, int n_in,
                              void* d_out, int out_size, void* d_ws,
                              size_t ws_size, hipStream_t stream) {
  (void)in_sizes; (void)n_in; (void)out_size; (void)ws_size;
  const int M = BATCH * SEQ_T;        // 3584
  const int Mxa = BATCH * SEQ_XA;     // 12000
  const int D = D_MODEL, F = F_MLP;

  const float* x_in  = (const float*)d_in[0];
  const float* xa_in = (const float*)d_in[1];
  const float* aln_g = (const float*)d_in[4];
  const float* aln_b = (const float*)d_in[5];
  const float* Wq = (const float*)d_in[6];   const float* bq = (const float*)d_in[7];
  const float* Wk = (const float*)d_in[8];
  const float* Wv = (const float*)d_in[9];   const float* bv = (const float*)d_in[10];
  const float* Wo = (const float*)d_in[11];  const float* bo = (const float*)d_in[12];
  const float* cln_g = (const float*)d_in[13];
  const float* cln_b = (const float*)d_in[14];
  const float* cWq = (const float*)d_in[15]; const float* cbq = (const float*)d_in[16];
  const float* cWk = (const float*)d_in[17];
  const float* cWv = (const float*)d_in[18]; const float* cbv = (const float*)d_in[19];
  const float* cWo = (const float*)d_in[20]; const float* cbo = (const float*)d_in[21];
  const float* mln_g = (const float*)d_in[22];
  const float* mln_b = (const float*)d_in[23];
  const float* W1 = (const float*)d_in[24];  const float* b1 = (const float*)d_in[25];
  const float* W2 = (const float*)d_in[26];  const float* b2 = (const float*)d_in[27];

  float* out_x = (float*)d_out;
  float* out_k = out_x + (size_t)M * D;
  float* out_v = out_k + (size_t)M * D;

  // ---- workspace carve ----
  char* wp = (char*)d_ws;
  auto take = [&](size_t bytes) { char* p = wp; wp += alignup(bytes); return p; };
  float*          xcur  = (float*)         take((size_t)M * D * 4);
  unsigned short* h_bf  = (unsigned short*)take((size_t)M * D * 2);
  unsigned short* q_bf  = (unsigned short*)take((size_t)M * D * 2);
  unsigned short* k_bf  = (unsigned short*)take((size_t)M * D * 2);
  unsigned short* v_bf  = (unsigned short*)take((size_t)M * D * 2);
  unsigned short* wv_bf = (unsigned short*)take((size_t)M * D * 2);
  unsigned short* xa_bf = (unsigned short*)take((size_t)Mxa * D * 2);
  unsigned short* ck_bf = (unsigned short*)take((size_t)Mxa * D * 2);
  unsigned short* cv_bf = (unsigned short*)take((size_t)Mxa * D * 2);
  unsigned short* hid_bf= (unsigned short*)take((size_t)M * F * 2);
  unsigned short* wWq = (unsigned short*)take((size_t)D * D * 2);
  unsigned short* wWk = (unsigned short*)take((size_t)D * D * 2);
  unsigned short* wWv = (unsigned short*)take((size_t)D * D * 2);
  unsigned short* wWo = (unsigned short*)take((size_t)D * D * 2);
  unsigned short* wcWq= (unsigned short*)take((size_t)D * D * 2);
  unsigned short* wcWk= (unsigned short*)take((size_t)D * D * 2);
  unsigned short* wcWv= (unsigned short*)take((size_t)D * D * 2);
  unsigned short* wcWo= (unsigned short*)take((size_t)D * D * 2);
  unsigned short* wW1 = (unsigned short*)take((size_t)D * F * 2);
  unsigned short* wW2 = (unsigned short*)take((size_t)F * D * 2);

  auto conv = [&](const float* s, unsigned short* d, size_t n) {
    int blocks = (int)((n + 2047) / 2048);
    if (blocks > 8192) blocks = 8192;
    f32_to_bf16_kernel<<<blocks, 256, 0, stream>>>(s, d, n);
  };
  conv(Wq, wWq, (size_t)D * D);   conv(Wk, wWk, (size_t)D * D);
  conv(Wv, wWv, (size_t)D * D);   conv(Wo, wWo, (size_t)D * D);
  conv(cWq, wcWq, (size_t)D * D); conv(cWk, wcWk, (size_t)D * D);
  conv(cWv, wcWv, (size_t)D * D); conv(cWo, wcWo, (size_t)D * D);
  conv(W1, wW1, (size_t)D * F);   conv(W2, wW2, (size_t)F * D);
  conv(xa_in, xa_bf, (size_t)Mxa * D);

  auto gemm = [&](const unsigned short* A, const unsigned short* Bm,
                  const float* bias, const float* res, float* oF,
                  unsigned short* oB, int Mr, int Nc, int Kc, int epi) {
    dim3 g((Mr + GBM - 1) / GBM, Nc / GBN);
    switch (epi) {
      case EPI_B16:
        gemm_bf16_kernel<EPI_B16><<<g, 256, 0, stream>>>(A, Bm, bias, res, oF, oB, Mr, Nc, Kc);
        break;
      case EPI_F32 | EPI_B16:
        gemm_bf16_kernel<EPI_F32 | EPI_B16><<<g, 256, 0, stream>>>(A, Bm, bias, res, oF, oB, Mr, Nc, Kc);
        break;
      case EPI_F32 | EPI_RES:
        gemm_bf16_kernel<EPI_F32 | EPI_RES><<<g, 256, 0, stream>>>(A, Bm, bias, res, oF, oB, Mr, Nc, Kc);
        break;
      case EPI_GELU | EPI_B16:
        gemm_bf16_kernel<EPI_GELU | EPI_B16><<<g, 256, 0, stream>>>(A, Bm, bias, res, oF, oB, Mr, Nc, Kc);
        break;
      default: break;
    }
  };

  const float scale = 0.125f;  // hd^-0.5
  dim3 agrid((SEQ_T + 127) / 128, N_HEADS, BATCH);

  // ---- self-attention ----
  layernorm_kernel<<<M, 256, 0, stream>>>(x_in, aln_g, aln_b, h_bf);
  gemm(h_bf, wWq, bq, nullptr, nullptr, q_bf, M, D, D, EPI_B16);
  gemm(h_bf, wWk, nullptr, nullptr, out_k, k_bf, M, D, D, EPI_F32 | EPI_B16);
  gemm(h_bf, wWv, bv, nullptr, out_v, v_bf, M, D, D, EPI_F32 | EPI_B16);
  attn_kernel<<<agrid, 256, 0, stream>>>(q_bf, k_bf, v_bf, wv_bf,
                                         SEQ_T, SEQ_T, 1, scale);
  gemm(wv_bf, wWo, bo, x_in, xcur, nullptr, M, D, D, EPI_F32 | EPI_RES);

  // ---- cross-attention ----
  layernorm_kernel<<<M, 256, 0, stream>>>(xcur, cln_g, cln_b, h_bf);
  gemm(h_bf, wcWq, cbq, nullptr, nullptr, q_bf, M, D, D, EPI_B16);
  gemm(xa_bf, wcWk, nullptr, nullptr, nullptr, ck_bf, Mxa, D, D, EPI_B16);
  gemm(xa_bf, wcWv, cbv, nullptr, nullptr, cv_bf, Mxa, D, D, EPI_B16);
  attn_kernel<<<agrid, 256, 0, stream>>>(q_bf, ck_bf, cv_bf, wv_bf,
                                         SEQ_T, SEQ_XA, 0, scale);
  gemm(wv_bf, wcWo, cbo, xcur, xcur, nullptr, M, D, D, EPI_F32 | EPI_RES);

  // ---- MLP (exact GELU) ----
  layernorm_kernel<<<M, 256, 0, stream>>>(xcur, mln_g, mln_b, h_bf);
  gemm(h_bf, wW1, b1, nullptr, nullptr, hid_bf, M, F, D, EPI_GELU | EPI_B16);
  gemm(hid_bf, wW2, b2, xcur, out_x, nullptr, M, D, F, EPI_F32 | EPI_RES);
}
